// SSAModule_17059610100246
// MI455X (gfx1250) — matrix operand order repaired and checked by reading, not compiled, not run on hardware
//
#include <hip/hip_runtime.h>
#include <hip/hip_bf16.h>

typedef __bf16 bf16_t;
typedef __attribute__((ext_vector_type(16))) __bf16 v16bf;
typedef __attribute__((ext_vector_type(8)))  __bf16 v8bf;
typedef __attribute__((ext_vector_type(8)))  float  v8f;
typedef __attribute__((ext_vector_type(4)))  float  v4f;

#define B_     16
#define C_     64
#define R_     32
#define T_     256
#define M_     2048     /* C_*R_ */
#define BM_    32768    /* B_*M_ */
#define INTER_ 16

static __device__ inline unsigned laneid() { return threadIdx.x & 31u; }

// ---- CDNA5 async global->LDS copy (ASYNCcnt-tracked) -----------------------
// 16 bytes per lane: LDS[lds_byte] = MEM[gaddr]
static __device__ inline void async_copy_b128(unsigned lds_byte, const void* gaddr) {
    asm volatile("global_load_async_to_lds_b128 %0, %1, off"
                 :: "v"(lds_byte), "v"(gaddr)
                 : "memory");
}
static __device__ inline void wait_async0() {
    asm volatile("s_wait_asynccnt 0" ::: "memory");
}

// ---- WMMA fragment helpers (wave32, 16x16x32 bf16, f32 accum) --------------
// A: 16x32 (MxK). lanes 0-15: row M=lane, elems 0..7 -> K 0..7, 8..15 -> K 16..23
//                 lanes 16-31: row M=lane-16, elems 0..7 -> K 8..15, 8..15 -> K 24..31
static __device__ inline v16bf load_a_f32(const float* __restrict__ src, int ld,
                                          int row0, int k0) {
    unsigned l = laneid();
    int m = row0 + (int)(l & 15u);
    int g = (l < 16u) ? 0 : 8;
    const float* p = src + (size_t)m * ld + k0 + g;
    v4f f0 = *(const v4f*)(p);
    v4f f1 = *(const v4f*)(p + 4);
    v4f f2 = *(const v4f*)(p + 16);
    v4f f3 = *(const v4f*)(p + 20);
    v16bf a;
#pragma unroll
    for (int j = 0; j < 4; ++j) {
        a[j]      = (bf16_t)f0[j];
        a[4 + j]  = (bf16_t)f1[j];
        a[8 + j]  = (bf16_t)f2[j];
        a[12 + j] = (bf16_t)f3[j];
    }
    return a;
}

static __device__ inline v16bf load_a_bf16(const bf16_t* src, int ld, int row0, int k0) {
    unsigned l = laneid();
    int m = row0 + (int)(l & 15u);
    int g = (l < 16u) ? 0 : 8;
    const bf16_t* p = src + (size_t)m * ld + k0 + g;
    v8bf lo = *(const v8bf*)(p);
    v8bf hi = *(const v8bf*)(p + 16);
    v16bf a;
#pragma unroll
    for (int j = 0; j < 8; ++j) { a[j] = lo[j]; a[8 + j] = hi[j]; }
    return a;
}

// B: 32x16 (KxN). Element (k, n) lives at src[n*ld + k] (contiguous in K per col).
static __device__ inline v16bf load_b_bf16(const bf16_t* src, int ld, int col0, int k0) {
    unsigned l = laneid();
    int n = col0 + (int)(l & 15u);
    int g = (l < 16u) ? 0 : 8;
    const bf16_t* p = src + (size_t)n * ld + k0 + g;
    v8bf lo = *(const v8bf*)(p);
    v8bf hi = *(const v8bf*)(p + 16);
    v16bf b;
#pragma unroll
    for (int j = 0; j < 8; ++j) { b[j] = lo[j]; b[8 + j] = hi[j]; }
    return b;
}

static __device__ inline v8f wmma_bf16(v16bf a, v16bf b, v8f c) {
    return __builtin_amdgcn_wmma_f32_16x16x32_bf16(
        false, a, false, b, (short)0, c, false, false);
}

// ---- K0: DCT-II ortho basis, stored bf16 row-major D[k][n] -----------------
__global__ void dct_basis_kernel(bf16_t* __restrict__ D) {
    int k = blockIdx.x;
    int n = threadIdx.x;
    float scale = (k == 0) ? 0.0625f : 0.08838834764831845f; // sqrt(1/256), sqrt(2/256)
    float v = scale * cosf(3.14159265358979323846f * (2.0f * n + 1.0f) * k / 512.0f);
    D[(size_t)k * T_ + n] = (bf16_t)v;
}

// ---- K1: Q = x @ D^T (rows b*M+m over 32768, cols k over 256) --------------
// D block tile (32 cols x 256) staged in LDS via async copy; K-loop double-buffered.
__global__ __launch_bounds__(256) void dct_gemm_kernel(
        const float* __restrict__ x, const bf16_t* __restrict__ D,
        bf16_t* __restrict__ Q, bf16_t* __restrict__ Qt) {
    __shared__ bf16_t sD[32 * T_];                     // 16 KB
    unsigned w = threadIdx.x >> 5;                     // 8 waves: 4 row x 2 col tiles
    int row0 = blockIdx.x * 64 + (int)(w >> 1) * 16;
    int colB = blockIdx.y * 32;
    int colL = (int)(w & 1) * 16;

    // stage D tile: 1024 x 16B units, 4 per thread
#pragma unroll
    for (int i = 0; i < 4; ++i) {
        int u = (int)threadIdx.x + i * 256;
        int cc = u >> 5;                  // 32 units (512 B) per column-row
        int no = (u & 31) * 8;
        async_copy_b128((unsigned)(size_t)&sD[0] + (unsigned)u * 16u,
                        D + (size_t)(colB + cc) * T_ + no);
    }
    wait_async0();
    __syncthreads();

    v8f c = {};
    v16bf a = load_a_f32(x, T_, row0, 0);
    v16bf b = load_b_bf16(sD, T_, colL, 0);
#pragma unroll
    for (int k0 = 32; k0 < T_; k0 += 32) {
        v16bf an = load_a_f32(x, T_, row0, k0);
        v16bf bn = load_b_bf16(sD, T_, colL, k0);
        c = wmma_bf16(a, b, c);
        a = an; b = bn;
    }
    c = wmma_bf16(a, b, c);

    unsigned l = laneid();
    int col   = colB + colL + (int)(l & 15u);
    int rbase = row0 + ((l < 16u) ? 0 : 8);
#pragma unroll
    for (int v = 0; v < 8; ++v) {
        int row = rbase + v;
        int bb  = row >> 11;
        int m   = row & 2047;
        bf16_t val = (bf16_t)c[v];
        Q[(size_t)row * T_ + col]            = val;
        Qt[((size_t)bb * T_ + col) * M_ + m] = val;
    }
}

// ---- K2: scores = Qt Qt^T / 16 ; softmax rows ; attn bf16 ------------------
// Shared A tile (16 t-rows x 2048 m) staged in LDS in two 32 KB halves.
__global__ __launch_bounds__(256) void attn_kernel(
        const bf16_t* __restrict__ Qt, bf16_t* __restrict__ attn) {
    __shared__ bf16_t sA[16 * 1024];                   // 32 KB half-tile
    __shared__ float  sc[16][T_];                      // 16 KB
    int b  = blockIdx.x;
    int t0 = blockIdx.y * 16;
    unsigned w = threadIdx.x >> 5;
    const bf16_t* Qtb = Qt + (size_t)b * T_ * M_;
    int s0 = (int)w * 32;
    v8f c0 = {}, c1 = {};

    for (int half = 0; half < 2; ++half) {
        __syncthreads();                               // prior half fully consumed
        // stage 16 rows x 1024 m : 2048 x 16B units, 8 per thread
#pragma unroll
        for (int i = 0; i < 8; ++i) {
            int u = (int)threadIdx.x + i * 256;
            int row = u >> 7;                          // 128 units (2 KB) per row
            int mo  = (u & 127) * 8;
            async_copy_b128((unsigned)(size_t)&sA[0] + (unsigned)u * 16u,
                            Qtb + (size_t)(t0 + row) * M_ + half * 1024 + mo);
        }
        wait_async0();
        __syncthreads();

        int kb = half * 1024;
        v16bf a  = load_a_bf16(sA, 1024, 0, 0);
        v16bf b0 = load_b_bf16(Qtb, M_, s0, kb);
        v16bf b1 = load_b_bf16(Qtb, M_, s0 + 16, kb);
        for (int kl = 32; kl < 1024; kl += 32) {
            v16bf an  = load_a_bf16(sA, 1024, 0, kl);
            v16bf bn0 = load_b_bf16(Qtb, M_, s0, kb + kl);
            v16bf bn1 = load_b_bf16(Qtb, M_, s0 + 16, kb + kl);
            c0 = wmma_bf16(a, b0, c0);
            c1 = wmma_bf16(a, b1, c1);
            a = an; b0 = bn0; b1 = bn1;
        }
        c0 = wmma_bf16(a, b0, c0);
        c1 = wmma_bf16(a, b1, c1);
    }

    unsigned l = laneid();
    int cn = (int)(l & 15u);
    int rb = (l < 16u) ? 0 : 8;
#pragma unroll
    for (int v = 0; v < 8; ++v) {
        sc[rb + v][s0 + cn]      = c0[v] * 0.0625f;    // 1/sqrt(256)
        sc[rb + v][s0 + 16 + cn] = c1[v] * 0.0625f;
    }
    __syncthreads();
    if (threadIdx.x < 16) {
        int r = (int)threadIdx.x;
        float mx = -3.4e38f;
        for (int s = 0; s < T_; ++s) mx = fmaxf(mx, sc[r][s]);
        float sum = 0.0f;
        for (int s = 0; s < T_; ++s) { float e = __expf(sc[r][s] - mx); sc[r][s] = e; sum += e; }
        float inv = 1.0f / sum;
        bf16_t* dst = attn + ((size_t)b * T_ + (t0 + r)) * T_;
        for (int s = 0; s < T_; ++s) dst[s] = (bf16_t)(sc[r][s] * inv);
    }
}

// ---- K3: att_out[b][m][t] = sum_s Q[b][m][s] * attn[b][t][s] ---------------
// attn block tile (32 t-cols x 256 s) staged in LDS via async copy.
__global__ __launch_bounds__(256) void attout_kernel(
        const bf16_t* __restrict__ Q, const bf16_t* __restrict__ attn,
        bf16_t* __restrict__ att_out) {
    __shared__ bf16_t sAt[32 * T_];                    // 16 KB
    int b = blockIdx.z;
    unsigned w = threadIdx.x >> 5;
    int m0 = blockIdx.x * 64 + (int)(w >> 1) * 16;
    int tB = blockIdx.y * 32;
    int tL = (int)(w & 1) * 16;
    const bf16_t* Qb = Q    + (size_t)b * M_ * T_;
    const bf16_t* At = attn + (size_t)b * T_ * T_;

#pragma unroll
    for (int i = 0; i < 4; ++i) {
        int u = (int)threadIdx.x + i * 256;
        int cc = u >> 5;
        int so = (u & 31) * 8;
        async_copy_b128((unsigned)(size_t)&sAt[0] + (unsigned)u * 16u,
                        At + (size_t)(tB + cc) * T_ + so);
    }
    wait_async0();
    __syncthreads();

    v8f c = {};
    v16bf a = load_a_bf16(Qb, T_, m0, 0);
    v16bf bb = load_b_bf16(sAt, T_, tL, 0);
#pragma unroll
    for (int s0 = 32; s0 < T_; s0 += 32) {
        v16bf an  = load_a_bf16(Qb, T_, m0, s0);
        v16bf bbn = load_b_bf16(sAt, T_, tL, s0);
        c = wmma_bf16(a, bb, c);
        a = an; bb = bbn;
    }
    c = wmma_bf16(a, bb, c);

    unsigned l = laneid();
    int col   = tB + tL + (int)(l & 15u);
    int rbase = m0 + ((l < 16u) ? 0 : 8);
#pragma unroll
    for (int v = 0; v < 8; ++v)
        att_out[((size_t)b * M_ + rbase + v) * T_ + col] = (bf16_t)c[v];
}

// ---- K4: out = x * sigmoid(W2[:64] @ relu(W1 @ att_out + b1) + b2) ---------
__global__ __launch_bounds__(256) void gate_kernel(
        const float* __restrict__ x, const bf16_t* __restrict__ att_out,
        const float* __restrict__ W1, const float* __restrict__ b1,
        const float* __restrict__ W2, const float* __restrict__ b2,
        float* __restrict__ out) {
    __shared__ float sW1[INTER_ * C_];
    __shared__ float sW2[C_ * INTER_];
    __shared__ float sb1[INTER_], sb2[C_];
    // async-stage both weight matrices (16 B per thread each)
    async_copy_b128((unsigned)(size_t)&sW1[threadIdx.x * 4], W1 + threadIdx.x * 4);
    async_copy_b128((unsigned)(size_t)&sW2[threadIdx.x * 4], W2 + threadIdx.x * 4);
    if (threadIdx.x < INTER_) sb1[threadIdx.x] = b1[threadIdx.x];
    if (threadIdx.x < C_)     sb2[threadIdx.x] = b2[threadIdx.x];
    wait_async0();
    __syncthreads();

    int pix = blockIdx.x * blockDim.x + threadIdx.x;   // over B*R*T = 131072
    int t = pix & (T_ - 1);
    int r = (pix >> 8) & (R_ - 1);
    int b = pix >> 13;
    size_t base = (((size_t)b * C_) * R_ + r) * T_ + t;

    float h[INTER_];
#pragma unroll
    for (int i = 0; i < INTER_; ++i) h[i] = sb1[i];
    for (int cc = 0; cc < C_; ++cc) {
        float a = (float)att_out[base + (size_t)cc * (R_ * T_)];
#pragma unroll
        for (int i = 0; i < INTER_; ++i) h[i] += sW1[i * C_ + cc] * a;
    }
#pragma unroll
    for (int i = 0; i < INTER_; ++i) h[i] = fmaxf(h[i], 0.0f);

    for (int c = 0; c < C_; ++c) {
        float z = sb2[c];
#pragma unroll
        for (int i = 0; i < INTER_; ++i) z += sW2[c * INTER_ + i] * h[i];
        float g = 1.0f / (1.0f + __expf(-z));
        size_t idx = base + (size_t)c * (R_ * T_);
        out[idx] = x[idx] * g;
    }
}

extern "C" void kernel_launch(void* const* d_in, const int* in_sizes, int n_in,
                              void* d_out, int out_size, void* d_ws, size_t ws_size,
                              hipStream_t stream) {
    (void)in_sizes; (void)n_in; (void)out_size; (void)ws_size;
    const float* x  = (const float*)d_in[0];
    const float* W1 = (const float*)d_in[1];
    const float* b1 = (const float*)d_in[2];
    const float* W2 = (const float*)d_in[3];
    const float* b2 = (const float*)d_in[4];
    float* out = (float*)d_out;

    char* ws = (char*)d_ws;
    const size_t szD    = (size_t)T_ * T_ * sizeof(bf16_t);       //   128 KB
    const size_t szQ    = (size_t)BM_ * T_ * sizeof(bf16_t);      //  16 MB
    const size_t szAttn = (size_t)B_ * T_ * T_ * sizeof(bf16_t);  //   2 MB
    bf16_t* D    = (bf16_t*)(ws);
    bf16_t* Q    = (bf16_t*)(ws + szD);
    bf16_t* Qt   = (bf16_t*)(ws + szD + szQ);
    bf16_t* attn = (bf16_t*)(ws + szD + 2 * szQ);
    bf16_t* att  = (bf16_t*)(ws + szD + 2 * szQ + szAttn);

    dct_basis_kernel<<<T_, T_, 0, stream>>>(D);
    dct_gemm_kernel<<<dim3(BM_ / 64, T_ / 32), 256, 0, stream>>>(x, D, Q, Qt);
    attn_kernel<<<dim3(B_, T_ / 16), 256, 0, stream>>>(Qt, attn);
    attout_kernel<<<dim3(M_ / 64, T_ / 32, B_), 256, 0, stream>>>(Q, attn, att);
    gate_kernel<<<(B_ * R_ * T_) / 256, 256, 0, stream>>>(x, att, W1, b1, W2, b2, out);
}
